// Atten_Head_8993661518267
// MI455X (gfx1250) — matrix operand order repaired and checked by reading, not compiled, not run on hardware
//
#include <hip/hip_runtime.h>
#include <hip/hip_bf16.h>

typedef __attribute__((ext_vector_type(16))) _Float16 v16h;
typedef __attribute__((ext_vector_type(8)))  _Float16 v8h;
typedef __attribute__((ext_vector_type(8)))  float    v8f;
typedef __attribute__((ext_vector_type(4)))  float    v4f;

#define BATCH 8
#define SEQL  2048
#define DIN   512
#define DOUT  64

// Workspace layout (units: _Float16 elements)
#define OWT  ((size_t)0)                    // W^T f16: 3 x [64][512]      = 98304
#define OEH  ((size_t)98304)                // E   f16: [L+64][64]         = 135168
#define OQH  ((size_t)233472)               // Q   f16: [B*L][64]          = 1048576
#define OKH  ((size_t)1282048)              // K   f16: [B*L][64]          = 1048576
#define OVT  ((size_t)2330624)               // V^T f16: [B][64][L]        = 1048576
// total = 3379200 halves = 6.45 MiB

__device__ __forceinline__ v8f wmma16(v16h a, v16h b, v8f c) {
  // D = A(16x32 f16) * B(32x16 f16) + C(16x16 f32)
  return __builtin_amdgcn_wmma_f32_16x16x32_f16(false, a, false, b,
                                                (short)0, c, false, false);
}

// A-operand frag (16x32 f16). ISA layout: lanes 0-15 hold row M=lane with
// K = {0..7, 16..23}; lanes 16-31 hold M=lane-16 with K = {8..15, 24..31}.
// Source is row-major f16 with rowStride halves.
__device__ __forceinline__ v16h load_a_frag(const _Float16* p, int rowStride, int lane) {
  int m  = lane & 15;
  int b8 = (lane >> 4) * 8;
  const _Float16* rp = p + (size_t)m * rowStride + b8;
  v8h lo = *(const v8h*)(rp);
  v8h hi = *(const v8h*)(rp + 16);
  v16h r;
#pragma unroll
  for (int i = 0; i < 8; ++i) { r[i] = lo[i]; r[i + 8] = hi[i]; }
  return r;
}

// Same A layout but converting from f32 source.
__device__ __forceinline__ v16h load_a_frag_f32(const float* p, int rowStride, int lane) {
  int m  = lane & 15;
  int b8 = (lane >> 4) * 8;
  const float* rp = p + (size_t)m * rowStride + b8;
  v4f a0 = *(const v4f*)(rp);
  v4f a1 = *(const v4f*)(rp + 4);
  v4f b0 = *(const v4f*)(rp + 16);
  v4f b1 = *(const v4f*)(rp + 20);
  v16h r;
#pragma unroll
  for (int i = 0; i < 4; ++i) {
    r[i]      = (_Float16)a0[i];
    r[i + 4]  = (_Float16)a1[i];
    r[i + 8]  = (_Float16)b0[i];
    r[i + 12] = (_Float16)b1[i];
  }
  return r;
}

// B-operand frag (32x16 f16), B[k][n] = M[n][k] where M is row-major f16.
// Layout: lanes 0-15 hold col n=lane, K={0..7, 16..23};
// lanes 16-31 hold n=lane-16, K={8..15, 24..31}.  -> two 16B loads per lane.
__device__ __forceinline__ v16h load_b_frag(const _Float16* p, int rowStride, int lane) {
  int n  = lane & 15;
  int kb = (lane >> 4) * 8;
  const _Float16* rp = p + (size_t)n * rowStride + kb;
  v8h lo = *(const v8h*)(rp);
  v8h hi = *(const v8h*)(rp + 16);
  v16h r;
#pragma unroll
  for (int i = 0; i < 8; ++i) { r[i] = lo[i]; r[i + 8] = hi[i]; }
  return r;
}

// ---------------------------------------------------------------- prep ----
// W^T (f32 [512][64] -> f16 [64][512]) for Q,K,V; E -> f16 with 64 zero-pad rows.
__global__ void prep_kernel(const float* __restrict__ WK, const float* __restrict__ WV,
                            const float* __restrict__ WQ, const float* __restrict__ E,
                            _Float16* __restrict__ ws) {
  int idx = blockIdx.x * blockDim.x + threadIdx.x;   // 456*512 = 233472 threads exactly
  if (idx < 98304) {
    int z   = idx >> 15;           // 0=Q, 1=K, 2=V
    int rem = idx & 32767;
    int n   = rem >> 9;
    int k   = rem & 511;
    const float* W = (z == 0) ? WQ : (z == 1) ? WK : WV;
    ws[OWT + idx] = (_Float16)W[k * 64 + n];
  } else {
    int i2 = idx - 98304;          // < (L+64)*64
    int e  = i2 >> 6;
    float v = (e < SEQL) ? E[i2] : 0.0f;
    ws[OEH + i2] = (_Float16)v;
  }
}

// ---------------------------------------------------------- projections ----
// One wave per 16x64 output tile of [B*L, 64]: the X tile (16x512 f32, the
// dominant HBM traffic of the whole problem) is loaded exactly ONCE and reused
// across all four 16-column weight groups (4 WMMAs per k-step, 64 per block).
// z: 0=Q (row-major), 1=K (row-major), 2=V (transposed per batch Vt[b][feat][l]).
__global__ __launch_bounds__(32) void proj_kernel(const float* __restrict__ Xk,
                                                  const float* __restrict__ Xv,
                                                  const float* __restrict__ Xq,
                                                  _Float16* __restrict__ ws) {
  int lane = threadIdx.x;
  int tile = blockIdx.x;           // row tile of B*L (1024 tiles)
  int z    = blockIdx.y;           // 0=Q 1=K 2=V
  const float*    X  = (z == 0) ? Xq : (z == 1) ? Xk : Xv;
  const _Float16* WT = ws + OWT + (size_t)z * 32768;
  const float*    Xt = X + (size_t)tile * 16 * DIN;

  v8f acc[4] = {{}, {}, {}, {}};
#pragma unroll 2
  for (int s = 0; s < 16; ++s) {
    v16h a = load_a_frag_f32(Xt + s * 32, DIN, lane);   // one A load ...
#pragma unroll
    for (int g = 0; g < 4; ++g) {                       // ... four WMMAs
      v16h b = load_b_frag(WT + (size_t)g * 16 * DIN + s * 32, DIN, lane);
      acc[g] = wmma16(a, b, acc[g]);
    }
  }

  int n  = lane & 15;
  int hl = lane >> 4;
#pragma unroll
  for (int g = 0; g < 4; ++g) {
#pragma unroll
    for (int r = 0; r < 8; ++r) {
      int row = tile * 16 + r + 8 * hl;        // C layout: lane holds (row, col n)
      _Float16 v = (_Float16)acc[g][r];
      if (z == 0)      ws[OQH + (size_t)row * 64 + g * 16 + n] = v;
      else if (z == 1) ws[OKH + (size_t)row * 64 + g * 16 + n] = v;
      else {
        int bb = row >> 11, ll = row & 2047;
        ws[OVT + (size_t)bb * 64 * SEQL + (size_t)(g * 16 + n) * SEQL + ll] = v;
      }
    }
  }
}

// ------------------------------------------------------ fused attention ----
// One wave = one 16-query tile; flash-attention loop over 32-key tiles.
// S_rel[q,k] = Q[q] . E[L-1-q+k]  (skew) is realized as 6 extra WMMAs over a
// 48-row E window + a per-VGPR lane permute (__shfl) in the C layout.
__global__ __launch_bounds__(256) void attn_kernel(const _Float16* __restrict__ ws,
                                                   float* __restrict__ out) {
  __shared__ _Float16 pbuf[8][16 * 32];        // per-wave 1KB P re-layout slab

  int lane = threadIdx.x & 31;
  int wid  = threadIdx.x >> 5;
  int b    = blockIdx.y;
  int q0   = (blockIdx.x * 8 + wid) * 16;

  const _Float16* Qh = ws + OQH + (size_t)b * SEQL * 64;
  const _Float16* Kh = ws + OKH + (size_t)b * SEQL * 64;
  const _Float16* Vt = ws + OVT + (size_t)b * 64 * SEQL;
  const _Float16* Eh = ws + OEH;

  // Query tile resident in registers: 16x64 = two A frags
  v16h qf0 = load_a_frag(Qh + (size_t)q0 * 64 + 0,  64, lane);
  v16h qf1 = load_a_frag(Qh + (size_t)q0 * 64 + 32, 64, lane);

  v8f o0 = {}, o1 = {}, o2 = {}, o3 = {};
  float mrow[8], lrow[8];
#pragma unroll
  for (int r = 0; r < 8; ++r) { mrow[r] = -1e30f; lrow[r] = 0.0f; }

  int n  = lane & 15;
  int hl = lane >> 4;
  const float scale = 0.022097086912079608f;   // 1/sqrt(2048)

  int ntiles = (q0 + 47) >> 5;
  for (int t = 0; t < ntiles; ++t) {
    int k0 = t * 32;

    // cover HBM latency for the next iteration's K/V tiles (global_prefetch_b8)
    if (t + 1 < ntiles) {
      __builtin_prefetch(Kh + (size_t)(k0 + 32) * 64 + (size_t)lane * 64, 0, 3);
      __builtin_prefetch(Vt + (size_t)(lane & 15) * SEQL * 4 + k0 + 32, 0, 3);
    }

    // ---- S = Q K^T  (16x32 scores: two 16-col groups) ----
    v8f s0 = {}, s1 = {};
    {
      v16h b00 = load_b_frag(Kh + (size_t)(k0)      * 64 + 0,  64, lane);
      v16h b01 = load_b_frag(Kh + (size_t)(k0)      * 64 + 32, 64, lane);
      s0 = wmma16(qf0, b00, s0); s0 = wmma16(qf1, b01, s0);
      v16h b10 = load_b_frag(Kh + (size_t)(k0 + 16) * 64 + 0,  64, lane);
      v16h b11 = load_b_frag(Kh + (size_t)(k0 + 16) * 64 + 32, 64, lane);
      s1 = wmma16(qf0, b10, s1); s1 = wmma16(qf1, b11, s1);
    }

    // ---- R = Q E_sub^T over 48 E rows starting at e0 ----
    int e0 = SEQL - 16 - q0 + k0;              // 0 <= e0 <= L-1; rows < L+48 (padded)
    v8f r0 = {}, r1 = {}, r2 = {};
    {
      v16h e00 = load_b_frag(Eh + (size_t)(e0)      * 64 + 0,  64, lane);
      v16h e01 = load_b_frag(Eh + (size_t)(e0)      * 64 + 32, 64, lane);
      r0 = wmma16(qf0, e00, r0); r0 = wmma16(qf1, e01, r0);
      v16h e10 = load_b_frag(Eh + (size_t)(e0 + 16) * 64 + 0,  64, lane);
      v16h e11 = load_b_frag(Eh + (size_t)(e0 + 16) * 64 + 32, 64, lane);
      r1 = wmma16(qf0, e10, r1); r1 = wmma16(qf1, e11, r1);
      v16h e20 = load_b_frag(Eh + (size_t)(e0 + 32) * 64 + 0,  64, lane);
      v16h e21 = load_b_frag(Eh + (size_t)(e0 + 32) * 64 + 32, 64, lane);
      r2 = wmma16(qf0, e20, r2); r2 = wmma16(qf1, e21, r2);
    }

    // ---- skew gather (lane permute), scale, causal mask ----
    float sv0[8], sv1[8];
#pragma unroll
    for (int r = 0; r < 8; ++r) {
      int m = r + 8 * hl;
      int j0 = n - m + 15;                      // [0,30] -> frag r0/r1
      int src0 = (j0 & 15) | (lane & 16);
      float ra = __shfl(r0[r], src0, 32);
      float rb = __shfl(r1[r], src0, 32);
      float rel0 = (j0 < 16) ? ra : rb;
      int j1 = n + 31 - m;                      // [16,46] -> frag r1/r2
      int src1 = (j1 & 15) | (lane & 16);
      float rc = __shfl(r1[r], src1, 32);
      float rd = __shfl(r2[r], src1, 32);
      float rel1 = (j1 < 32) ? rc : rd;
      float v0 = (s0[r] + rel0) * scale;
      float v1 = (s1[r] + rel1) * scale;
      if (k0 + n      > q0 + m) v0 = -1e30f;    // causal mask
      if (k0 + 16 + n > q0 + m) v1 = -1e30f;
      sv0[r] = v0; sv1[r] = v1;
    }

    // ---- online softmax (row = 16 lanes of one half-wave) ----
    float cexp[8];
#pragma unroll
    for (int r = 0; r < 8; ++r) {
      float mx = fmaxf(sv0[r], sv1[r]);
#pragma unroll
      for (int d = 1; d < 16; d <<= 1) mx = fmaxf(mx, __shfl_xor(mx, d, 32));
      float mnew = fmaxf(mrow[r], mx);
      float c  = __expf(mrow[r] - mnew);
      float p0 = __expf(sv0[r] - mnew);
      float p1 = __expf(sv1[r] - mnew);
      float sum = p0 + p1;
#pragma unroll
      for (int d = 1; d < 16; d <<= 1) sum += __shfl_xor(sum, d, 32);
      lrow[r] = lrow[r] * c + sum;
      mrow[r] = mnew;
      cexp[r] = c;
      sv0[r] = p0; sv1[r] = p1;
    }
#pragma unroll
    for (int r = 0; r < 8; ++r) {
      o0[r] *= cexp[r]; o1[r] *= cexp[r]; o2[r] *= cexp[r]; o3[r] *= cexp[r];
    }

    // ---- P (C layout) -> LDS -> A frag ----
    _Float16* pb = &pbuf[wid][0];
#pragma unroll
    for (int r = 0; r < 8; ++r) {
      int m = r + 8 * hl;
      pb[m * 32 + n]      = (_Float16)sv0[r];
      pb[m * 32 + 16 + n] = (_Float16)sv1[r];
    }
    __builtin_amdgcn_wave_barrier();
    asm volatile("s_wait_dscnt 0" ::: "memory");   // same-wave LDS RAW
    v16h pa = load_a_frag(pb, 32, lane);

    // ---- O += P @ V (V stored transposed: contiguous key runs per lane) ----
    o0 = wmma16(pa, load_b_frag(Vt + (size_t)0  * SEQL + k0, SEQL, lane), o0);
    o1 = wmma16(pa, load_b_frag(Vt + (size_t)16 * SEQL + k0, SEQL, lane), o1);
    o2 = wmma16(pa, load_b_frag(Vt + (size_t)32 * SEQL + k0, SEQL, lane), o2);
    o3 = wmma16(pa, load_b_frag(Vt + (size_t)48 * SEQL + k0, SEQL, lane), o3);
  }

  // ---- normalize and store [B, L, 64] f32 ----
#pragma unroll
  for (int r = 0; r < 8; ++r) {
    int m = r + 8 * hl;
    float inv = 1.0f / lrow[r];
    size_t base = ((size_t)b * SEQL + q0 + m) * 64 + n;
    out[base + 0]  = o0[r] * inv;
    out[base + 16] = o1[r] * inv;
    out[base + 32] = o2[r] * inv;
    out[base + 48] = o3[r] * inv;
  }
}

extern "C" void kernel_launch(void* const* d_in, const int* in_sizes, int n_in,
                              void* d_out, int out_size, void* d_ws, size_t ws_size,
                              hipStream_t stream) {
  const float* Xk = (const float*)d_in[0];
  const float* Xv = (const float*)d_in[1];
  const float* Xq = (const float*)d_in[2];
  const float* WK = (const float*)d_in[3];
  const float* WV = (const float*)d_in[4];
  const float* WQ = (const float*)d_in[5];
  const float* E  = (const float*)d_in[6];
  _Float16* ws = (_Float16*)d_ws;
  float* out = (float*)d_out;

  // 1) weight transpose + E conversion (233472 elements, exact fit)
  prep_kernel<<<dim3(456), 512, 0, stream>>>(WK, WV, WQ, E, ws);
  // 2) Q/K/V projections: 1024 row tiles x {Q,K,V}; X tile read once per wave
  proj_kernel<<<dim3(1024, 3), 32, 0, stream>>>(Xk, Xv, Xq, ws);
  // 3) fused rel-pos flash attention: 8 waves/block, 16 blocks x 8 batches
  attn_kernel<<<dim3(16, 8), 256, 0, stream>>>(ws, out);
}
